// MultiScaleIndexer_76982993814141
// MI455X (gfx1250) — matrix operand order repaired and checked by reading, not compiled, not run on hardware
//
#include <hip/hip_runtime.h>
#include <hip/hip_bf16.h>

typedef __attribute__((ext_vector_type(16))) _Float16 v16h;
typedef __attribute__((ext_vector_type(8)))  _Float16 v8h;
typedef __attribute__((ext_vector_type(4)))  _Float16 v4h;
typedef __attribute__((ext_vector_type(8)))  float    v8f;

#define D_MODEL 1024
#define T_SEQ   2048
#define B_BATCH 2
#define BT      (B_BATCH * T_SEQ)   // 4096 tokens
#define PW      816                 // padded projection width (801 real cols)

// Workspace layout (bytes):
//   xh : BT*D_MODEL f16   =  8,388,608
//   wc : PW*D_MODEL f16   =  1,671,168
//   P  : BT*PW     f32    = 13,369,344   (total ~23.4 MB)

// ---------------------------------------------------------------------------
// x (f32) -> xh (f16), 4 elements per thread
__global__ void msidx_cvt_x(const float* __restrict__ x, _Float16* __restrict__ xh) {
    int i = (blockIdx.x * blockDim.x + threadIdx.x) * 4;
    float4 v = *(const float4*)(x + i);
    v4h o;
    o[0] = (_Float16)v.x; o[1] = (_Float16)v.y;
    o[2] = (_Float16)v.z; o[3] = (_Float16)v.w;
    *(v4h*)(xh + i) = o;
}

// ---------------------------------------------------------------------------
// Concatenate all weight matrices (row-major [out, D]) into wc f16 [PW x D].
// Row map: [0,32) qw0 | [32,160) qw1 | [160,672) qw2 | [672,688) kw0 |
// [688,720) kw1 | [720,784) kw2 | [784,786) ww0 | [786,790) ww1 |
// [790,798) ww2 | [798,801) combw | [801,816) zero pad
__global__ void msidx_concat_w(const float* __restrict__ qw0, const float* __restrict__ kw0,
                               const float* __restrict__ ww0, const float* __restrict__ qw1,
                               const float* __restrict__ kw1, const float* __restrict__ ww1,
                               const float* __restrict__ qw2, const float* __restrict__ kw2,
                               const float* __restrict__ ww2, const float* __restrict__ combw,
                               _Float16* __restrict__ wc) {
    int idx = blockIdx.x * blockDim.x + threadIdx.x;
    if (idx >= PW * D_MODEL) return;
    int r = idx >> 10, c = idx & (D_MODEL - 1);
    const float* src = nullptr; int sr = 0;
    if      (r < 32)  { src = qw0;   sr = r;       }
    else if (r < 160) { src = qw1;   sr = r - 32;  }
    else if (r < 672) { src = qw2;   sr = r - 160; }
    else if (r < 688) { src = kw0;   sr = r - 672; }
    else if (r < 720) { src = kw1;   sr = r - 688; }
    else if (r < 784) { src = kw2;   sr = r - 720; }
    else if (r < 786) { src = ww0;   sr = r - 784; }
    else if (r < 790) { src = ww1;   sr = r - 786; }
    else if (r < 798) { src = ww2;   sr = r - 790; }
    else if (r < 801) { src = combw; sr = r - 798; }
    wc[idx] = src ? (_Float16)src[sr * D_MODEL + c] : (_Float16)0.f;
}

// ---------------------------------------------------------------------------
// Projection GEMM: P[BT x PW] (f32) = xh[BT x D] * wc[PW x D]^T via WMMA f16.
// One wave per 16x16 tile; 4 waves/block cover 4 consecutive N-tiles.
__global__ void msidx_proj_gemm(const _Float16* __restrict__ xh,
                                const _Float16* __restrict__ wc,
                                float* __restrict__ P) {
    int wid  = threadIdx.x >> 5;
    int lane = threadIdx.x & 31;
    int nt = blockIdx.x * 4 + wid;
    if (nt >= PW / 16) return;               // wave-uniform guard
    int mt = blockIdx.y;
    int lr = lane & 15, hf = lane >> 4;

    const _Float16* arow = xh + (size_t)(mt * 16 + lr) * D_MODEL;  // A: M = lane%16
    const _Float16* brow = wc + (size_t)(nt * 16 + lr) * D_MODEL;  // B: N = lane%16

    v8f c;
    #pragma unroll
    for (int v = 0; v < 8; v++) c[v] = 0.f;

    #pragma unroll 4
    for (int kb = 0; kb < D_MODEL; kb += 32) {
        // A fragment (16x32 f16): lane half hf -> K in [kb+16hf, +16), contiguous
        v8h a0 = *(const v8h*)(arow + kb + 16 * hf);
        v8h a1 = *(const v8h*)(arow + kb + 16 * hf + 8);
        v16h a;
        #pragma unroll
        for (int j = 0; j < 8; j++) { a[j] = a0[j]; a[8 + j] = a1[j]; }
        // B fragment (32x16 f16): lane half hf -> K in [kb+16hf, +16), contiguous
        v16h b = *(const v16h*)(brow + kb + 16 * hf);
        c = __builtin_amdgcn_wmma_f32_16x16x32_f16(false, a, false, b, (short)0, c, false, false);
    }

    // C layout: VGPR v -> row M = v + 8*hf, col N = lane%16
    #pragma unroll
    for (int v = 0; v < 8; v++)
        P[(size_t)(mt * 16 + v + 8 * hf) * PW + nt * 16 + lr] = c[v];
}

// ---------------------------------------------------------------------------
// Band attention: one wave per 16x16 output tile in the 143-wide causal band.
// grid = (9 s-tiles, T/16 t-tiles, B)
__global__ void msidx_band_attn(const float* __restrict__ P,
                                const float* __restrict__ combb,
                                float* __restrict__ out) {
    const int SC[3]  = {32, 64, 128};
    const int HH[3]  = {2, 4, 8};
    const int DDv[3] = {16, 32, 64};
    const int QO[3]  = {0, 32, 160};
    const int KO[3]  = {672, 688, 720};
    const int WO[3]  = {784, 786, 790};

    int j  = blockIdx.x;            // s-tile index within band, 0..8
    int tt = blockIdx.y;            // t-tile, 0..127
    int b  = blockIdx.z;
    int t0 = tt * 16;
    int s0 = t0 - 128 + j * 16;
    if (s0 < 0) return;             // tile entirely before sequence start

    int lane = threadIdx.x & 31;
    int lr = lane & 15, hf = lane >> 4;
    const float* Pb = P + (size_t)b * T_SEQ * PW;

    // per-row softmax over the 3 scale logits (P cols 798..800) + combb
    float cb0 = combb[0], cb1 = combb[1], cb2 = combb[2];
    float sw[3][8];
    #pragma unroll
    for (int v = 0; v < 8; v++) {
        const float* row = Pb + (size_t)(t0 + v + 8 * hf) * PW + 798;
        float l0 = row[0] + cb0, l1 = row[1] + cb1, l2 = row[2] + cb2;
        float mx = fmaxf(l0, fmaxf(l1, l2));
        float e0 = __expf(l0 - mx), e1 = __expf(l1 - mx), e2 = __expf(l2 - mx);
        float inv = 1.f / (e0 + e1 + e2);
        sw[0][v] = e0 * inv; sw[1][v] = e1 * inv; sw[2][v] = e2 * inv;
    }

    float acc[8];
    #pragma unroll
    for (int v = 0; v < 8; v++) acc[v] = 0.f;

    #pragma unroll
    for (int i = 0; i < 3; i++) {
        const int scale = SC[i];
        if ((t0 - s0) >= scale + 15) continue;   // tile outside this scale's band (uniform)
        const int d   = DDv[i];
        const int nch = (d + 31) / 32;           // K chunks of 32 (d=16 zero-padded)

        // B fragments: k vectors for the 16 s-columns of this tile
        v16h bf[2];
        const float* krow = Pb + (size_t)(s0 + lr) * PW + KO[i];
        #pragma unroll
        for (int ch = 0; ch < nch; ch++) {
            #pragma unroll
            for (int jj = 0; jj < 16; jj++) {
                int kk = 32 * ch + 16 * hf + jj;
                bf[ch][jj] = (kk < d) ? (_Float16)krow[kk] : (_Float16)0.f;
            }
        }

        float sc[8];
        #pragma unroll
        for (int v = 0; v < 8; v++) sc[v] = 0.f;

        #pragma unroll
        for (int hh = 0; hh < HH[i]; hh++) {
            const float* qrow = Pb + (size_t)(t0 + lr) * PW + QO[i] + hh * d;
            v8f dacc;
            #pragma unroll
            for (int v = 0; v < 8; v++) dacc[v] = 0.f;
            #pragma unroll
            for (int ch = 0; ch < nch; ch++) {
                v16h a;
                #pragma unroll
                for (int jj = 0; jj < 8; jj++) {
                    int k1 = 32 * ch + 16 * hf + jj;
                    a[jj]     = (k1 < d) ? (_Float16)qrow[k1] : (_Float16)0.f;
                    int k2 = 32 * ch + 16 * hf + 8 + jj;
                    a[8 + jj] = (k2 < d) ? (_Float16)qrow[k2] : (_Float16)0.f;
                }
                dacc = __builtin_amdgcn_wmma_f32_16x16x32_f16(false, a, false, bf[ch],
                                                              (short)0, dacc, false, false);
            }
            // relu(dot) * w[t, hh], sum over heads (f32)
            #pragma unroll
            for (int v = 0; v < 8; v++) {
                int t = t0 + v + 8 * hf;
                float wv = Pb[(size_t)t * PW + WO[i] + hh];
                sc[v] += fmaxf(dacc[v], 0.f) * wv;
            }
        }

        // band mask + softmax scale weight
        #pragma unroll
        for (int v = 0; v < 8; v++) {
            int t = t0 + v + 8 * hf;
            int s = s0 + lr;
            bool in = (s <= t) && (s > t - scale);
            acc[v] += in ? sw[i][v] * sc[v] : 0.f;
        }
    }

    #pragma unroll
    for (int v = 0; v < 8; v++) {
        int t = t0 + v + 8 * hf;
        out[((size_t)b * T_SEQ + t) * T_SEQ + s0 + lr] = acc[v];
    }
}

// ---------------------------------------------------------------------------
extern "C" void kernel_launch(void* const* d_in, const int* in_sizes, int n_in,
                              void* d_out, int out_size, void* d_ws, size_t ws_size,
                              hipStream_t stream) {
    const float* x     = (const float*)d_in[0];
    const float* qw0   = (const float*)d_in[1];
    const float* kw0   = (const float*)d_in[2];
    const float* ww0   = (const float*)d_in[3];
    const float* qw1   = (const float*)d_in[4];
    const float* kw1   = (const float*)d_in[5];
    const float* ww1   = (const float*)d_in[6];
    const float* qw2   = (const float*)d_in[7];
    const float* kw2   = (const float*)d_in[8];
    const float* ww2   = (const float*)d_in[9];
    const float* combw = (const float*)d_in[10];
    const float* combb = (const float*)d_in[11];

    char* ws = (char*)d_ws;
    _Float16* xh = (_Float16*)ws;
    _Float16* wc = (_Float16*)(ws + (size_t)BT * D_MODEL * 2);
    float*    P  = (float*)(ws + (size_t)BT * D_MODEL * 2 + (size_t)PW * D_MODEL * 2);

    // zero entire output; band kernel fills only the 143-wide causal band
    hipMemsetAsync(d_out, 0, (size_t)out_size * sizeof(float), stream);

    msidx_cvt_x<<<dim3((BT * D_MODEL) / (256 * 4)), dim3(256), 0, stream>>>(x, xh);
    msidx_concat_w<<<dim3((PW * D_MODEL + 255) / 256), dim3(256), 0, stream>>>(
        qw0, kw0, ww0, qw1, kw1, ww1, qw2, kw2, ww2, combw, wc);
    msidx_proj_gemm<<<dim3((PW / 16 + 3) / 4, BT / 16), dim3(128), 0, stream>>>(xh, wc, P);
    msidx_band_attn<<<dim3(9, T_SEQ / 16, B_BATCH), dim3(32), 0, stream>>>(P, combb, (float*)d_out);
}
